// ContentMultiheadAttention_51299089383478
// MI455X (gfx1250) — matrix operand order repaired and checked by reading, not compiled, not run on hardware
//
#include <hip/hip_runtime.h>
#include <hip/hip_bf16.h>
#include <stdint.h>

// ---------------------------------------------------------------------------
// MHA forward, L=512 B=32 E=1024 H=16 (hd=64), gfx1250 wave32 + WMMA.
//   K1: in-projection GEMM -> bf16 Q/K/V (head-split, Q pre-scaled, V^T)
//   K2: flash attention, 4 waves/block sharing LDS-staged K/V chunks
//       (staged with GLOBAL_LOAD_ASYNC_TO_LDS_B128, ASYNCcnt-tracked)
//   K3: out-projection GEMM (+async bf16 A staging) -> fp32 out
// All matmuls: v_wmma_f32_16x16x32_bf16 with fp32 accumulation.
// Fragments are preloaded in bulk so ds_load clauses overlap the WMMA pipe.
// ---------------------------------------------------------------------------

typedef __attribute__((ext_vector_type(16))) __bf16 bf16x16;
typedef __attribute__((ext_vector_type(8)))  float  f32x8;

union Frag16 { bf16x16 v; uint4 q[2]; };

__device__ __forceinline__ uint16_t f2bf(float f) {
  return (uint16_t)((__float_as_uint(f) + 0x8000u) >> 16);
}
__device__ __forceinline__ uint32_t pk2bf(float lo, float hi) {
  uint32_t a = __float_as_uint(lo) + 0x8000u;
  uint32_t b = __float_as_uint(hi) + 0x8000u;
  return __builtin_amdgcn_perm(b, a, 0x07060302u);   // {b[31:16], a[31:16]}
}
__device__ __forceinline__ uint4 cvt8(float4 a, float4 b) {
  return make_uint4(pk2bf(a.x,a.y), pk2bf(a.z,a.w), pk2bf(b.x,b.y), pk2bf(b.z,b.w));
}

// CDNA5 async global->LDS copy (16B per lane), tracked by ASYNCcnt.
__device__ __forceinline__ void async_copy16(void* lds_dst, const void* gsrc) {
  uint32_t l = (uint32_t)(uintptr_t)lds_dst;          // low 32 bits = LDS offset
  asm volatile("global_load_async_to_lds_b128 %0, %1, off"
               :: "v"(l), "v"((uint64_t)(uintptr_t)gsrc) : "memory");
}
__device__ __forceinline__ void async_wait0() {
  asm volatile("s_wait_asynccnt 0x0" ::: "memory");
}

#define LT 512
#define BT 32
#define ET 1024
#define HT 16
#define HD 64
#define NEGV (-1e9f)

#define BM 128
#define BN 128
#define KC 32
#define LDST 40    // padded LDS row stride (bf16 elems): 80B, 16B-aligned
#define KST 72     // K-chunk LDS stride (bf16 elems): 144B, 16B-aligned

// ---------------------------------------------------------------------------
// K1: in-projection.  Column block selects segment: 0=Q(Wq), 1=K(Wk), 2=V(Wv).
// ---------------------------------------------------------------------------
__global__ void __launch_bounds__(256)
in_proj_kernel(const float* __restrict__ Q, const float* __restrict__ K,
               const float* __restrict__ V, const float* __restrict__ W,
               const float* __restrict__ bias,
               uint16_t* __restrict__ qws,   // [B*H][L][hd] bf16 (pre-scaled)
               uint16_t* __restrict__ kws,   // [B*H][L][hd] bf16
               uint16_t* __restrict__ vws)   // [B*H][hd][L] bf16 (transposed)
{
  __shared__ uint16_t As[BM * LDST];
  __shared__ uint16_t Bs[BN * LDST];

  const int m0 = blockIdx.x * BM;
  const int n0 = blockIdx.y * BN;
  const int seg = n0 >> 10;
  const float* X = (seg == 0) ? Q : ((seg == 1) ? K : V);

  const int tid  = threadIdx.x;
  const int lane = tid & 31;
  const int lo   = lane & 15, hi = lane >> 4;
  const int wave = tid >> 5;
  const int wm = wave >> 1, wn = wave & 1;       // 4 waves in M, 2 in N

  f32x8 acc[2][4];
  for (int i = 0; i < 2; ++i) for (int j = 0; j < 4; ++j) acc[i][j] = (f32x8)0.0f;

  const int srow = tid >> 1;
  const int scol = (tid & 1) * 16;

  for (int k0 = 0; k0 < ET; k0 += KC) {
    { // stage A (fp32 -> bf16, v_perm pack)
      const float4* s4 = (const float4*)(X + (size_t)(m0 + srow) * ET + k0 + scol);
      uint4* dst = (uint4*)(As + srow * LDST + scol);
      dst[0] = cvt8(s4[0], s4[1]);
      dst[1] = cvt8(s4[2], s4[3]);
    }
    { // stage B = W rows (n-major => contiguous B-fragments)
      const float4* s4 = (const float4*)(W + (size_t)(n0 + srow) * ET + k0 + scol);
      uint4* dst = (uint4*)(Bs + srow * LDST + scol);
      dst[0] = cvt8(s4[0], s4[1]);
      dst[1] = cvt8(s4[2], s4[3]);
    }
    if (k0 + KC < ET) {   // global_prefetch_b8 for next K-slab
      __builtin_prefetch(X + (size_t)(m0 + srow) * ET + k0 + KC + scol, 0, 1);
      __builtin_prefetch(W + (size_t)(n0 + srow) * ET + k0 + KC + scol, 0, 1);
    }
    __syncthreads();

    // bulk-preload all fragments so the ds_load clause overlaps the WMMA pipe
    Frag16 afr[2], bfr[4];
    for (int mt = 0; mt < 2; ++mt) {
      const uint16_t* ap = As + (wm * 32 + mt * 16 + lo) * LDST;
      afr[mt].q[0] = *(const uint4*)(ap + hi * 8);       // K 0..7  / 8..15
      afr[mt].q[1] = *(const uint4*)(ap + 16 + hi * 8);  // K 16..23 / 24..31
    }
    for (int nt = 0; nt < 4; ++nt) {
      const uint16_t* bp = Bs + (wn * 64 + nt * 16 + lo) * LDST + hi * 8;
      bfr[nt].q[0] = *(const uint4*)(bp);
      bfr[nt].q[1] = *(const uint4*)(bp + 16);
    }
    for (int nt = 0; nt < 4; ++nt)
      for (int mt = 0; mt < 2; ++mt)
        acc[mt][nt] = __builtin_amdgcn_wmma_f32_16x16x32_bf16(
            false, afr[mt].v, false, bfr[nt].v, (short)0, acc[mt][nt], false, false);
    __syncthreads();
  }

  for (int nt = 0; nt < 4; ++nt) {
    const int ncol = n0 + wn * 64 + nt * 16 + lo;
    const float bsum = bias[ncol];
    const int nin = ncol & (ET - 1);
    const int h = nin >> 6, d = nin & 63;
    for (int mt = 0; mt < 2; ++mt) {
      for (int vg = 0; vg < 8; ++vg) {
        const int mrow = m0 + wm * 32 + mt * 16 + vg + hi * 8;
        const int l = mrow >> 5, b = mrow & 31;          // m = l*B + b
        const float val = acc[mt][nt][vg] + bsum;
        const size_t bh = (size_t)(b * HT + h);
        if (seg == 0)       qws[(bh * LT + l) * HD + d] = f2bf(val * 0.125f);
        else if (seg == 1)  kws[(bh * LT + l) * HD + d] = f2bf(val);
        else                vws[(bh * HD + d) * LT + l] = f2bf(val);
      }
    }
  }
}

// ---------------------------------------------------------------------------
// K2: flash attention.  128 threads = 4 waves, each wave owns one 16-row
// q-tile of the same (b,h); the K/V 32-key chunk is async-staged into LDS
// once per block and shared by all 4 waves.
// ---------------------------------------------------------------------------
__global__ void __launch_bounds__(128)
attn_kernel(const uint16_t* __restrict__ qws, const uint16_t* __restrict__ kws,
            const uint16_t* __restrict__ vws, const unsigned char* __restrict__ mask,
            uint16_t* __restrict__ ctx)       // [L*B][E] bf16
{
  __shared__ uint16_t Ks[32 * KST];           // K rows  [32 keys][64 d]
  __shared__ uint16_t Vs[64 * LDST];          // V^T rows [64 d][32 keys]
  __shared__ uint16_t Ps[4][16 * LDST];       // per-wave P tile

  const int tid  = threadIdx.x;
  const int lane = tid & 31;
  const int wave = tid >> 5;
  const int lo = lane & 15, hi = lane >> 4;

  const int bh = blockIdx.x >> 3;             // 0..511
  const int qg = blockIdx.x & 7;              // q-tile group
  const int b = bh >> 4, h = bh & 15;
  const int q0 = (qg * 4 + wave) * 16;

  const uint16_t* kbase = kws + (size_t)bh * LT * HD;
  const uint16_t* vbase = vws + (size_t)bh * HD * LT;
  const unsigned char* mbase = mask + ((size_t)b * LT + q0) * LT;

  // Q fragments (hd=64 -> 2 WMMA K-steps), loaded once per wave
  Frag16 qfr[2];
  {
    const uint16_t* qp = qws + ((size_t)bh * LT + q0 + lo) * HD;
    for (int ks = 0; ks < 2; ++ks) {
      qfr[ks].q[0] = *(const uint4*)(qp + ks * 32 + hi * 8);
      qfr[ks].q[1] = *(const uint4*)(qp + ks * 32 + 16 + hi * 8);
    }
  }

  f32x8 o[4];
  for (int nt = 0; nt < 4; ++nt) o[nt] = (f32x8)0.0f;
  float mrow[8], lrow[8];
  for (int vg = 0; vg < 8; ++vg) { mrow[vg] = -3.0e38f; lrow[vg] = 0.0f; }

  // staging coordinates (per thread, 16 bf16 = 32B each array)
  const int kr = tid >> 2, kcc = (tid & 3) * 16;   // K: 32 rows x 64 cols
  const int vr = tid >> 1, vcc = (tid & 1) * 16;   // V: 64 rows x 32 cols

  for (int kc = 0; kc < LT; kc += 32) {
    // async-stage K/V chunk into LDS (CDNA5 GLOBAL_LOAD_ASYNC_TO_LDS_B128)
    async_copy16(Ks + kr * KST + kcc,      kbase + (size_t)(kc + kr) * HD + kcc);
    async_copy16(Ks + kr * KST + kcc + 8,  kbase + (size_t)(kc + kr) * HD + kcc + 8);
    async_copy16(Vs + vr * LDST + vcc,     vbase + (size_t)vr * LT + kc + vcc);
    async_copy16(Vs + vr * LDST + vcc + 8, vbase + (size_t)vr * LT + kc + vcc + 8);
    async_wait0();
    __syncthreads();

    // S chunk = Q(16x64) @ K^T(64x32): preload 4 K-fragments, then 4 WMMAs
    Frag16 kf[2][2];
    for (int st = 0; st < 2; ++st) {
      const uint16_t* kp = Ks + (st * 16 + lo) * KST;
      for (int ks = 0; ks < 2; ++ks) {
        kf[st][ks].q[0] = *(const uint4*)(kp + ks * 32 + hi * 8);
        kf[st][ks].q[1] = *(const uint4*)(kp + ks * 32 + 16 + hi * 8);
      }
    }
    f32x8 s[2];
    for (int st = 0; st < 2; ++st) {
      s[st] = (f32x8)0.0f;
      for (int ks = 0; ks < 2; ++ks)
        s[st] = __builtin_amdgcn_wmma_f32_16x16x32_bf16(
            false, qfr[ks].v, false, kf[st][ks].v, (short)0, s[st], false, false);
    }
    // mask + online softmax (rows live across 16 lanes of one VGPR)
    float sc[8];
    for (int vg = 0; vg < 8; ++vg) {
      const int qr = vg + hi * 8;
      const unsigned char mk0 = mbase[(size_t)qr * LT + kc + lo];
      const unsigned char mk1 = mbase[(size_t)qr * LT + kc + 16 + lo];
      float s0 = mk0 ? NEGV : s[0][vg];
      float s1 = mk1 ? NEGV : s[1][vg];
      float t = fmaxf(s0, s1);
      for (int off = 1; off < 16; off <<= 1) t = fmaxf(t, __shfl_xor(t, off, 32));
      const float mn = fmaxf(mrow[vg], t);
      const float scale = __expf(mrow[vg] - mn);
      const float p0 = __expf(s0 - mn);
      const float p1 = __expf(s1 - mn);
      float rs = p0 + p1;
      for (int off = 1; off < 16; off <<= 1) rs += __shfl_xor(rs, off, 32);
      lrow[vg] = lrow[vg] * scale + rs;
      mrow[vg] = mn;
      sc[vg] = scale;
      Ps[wave][qr * LDST + lo]      = f2bf(p0);   // C-layout -> row-major
      Ps[wave][qr * LDST + 16 + lo] = f2bf(p1);
    }
    for (int nt = 0; nt < 4; ++nt)
      for (int vg = 0; vg < 8; ++vg) o[nt][vg] *= sc[vg];

    __syncthreads();                       // orders Ps store->load (in-wave)
    Frag16 pf, vf[4];                      // preload P + 4 V-fragments
    pf.q[0] = *(const uint4*)(Ps[wave] + lo * LDST + hi * 8);
    pf.q[1] = *(const uint4*)(Ps[wave] + lo * LDST + 16 + hi * 8);
    for (int nt = 0; nt < 4; ++nt) {
      const uint16_t* vp = Vs + (nt * 16 + lo) * LDST + hi * 8;
      vf[nt].q[0] = *(const uint4*)(vp);
      vf[nt].q[1] = *(const uint4*)(vp + 16);
    }
    for (int nt = 0; nt < 4; ++nt)         // O += P @ V (from LDS V^T)
      o[nt] = __builtin_amdgcn_wmma_f32_16x16x32_bf16(
          false, pf.v, false, vf[nt].v, (short)0, o[nt], false, false);
    __syncthreads();                       // protect LDS before re-staging
  }

  // normalize and scatter context back to [l,b,e] bf16
  for (int nt = 0; nt < 4; ++nt) {
    const int e = h * HD + nt * 16 + lo;
    for (int vg = 0; vg < 8; ++vg) {
      const int qr = q0 + vg + hi * 8;
      ctx[((size_t)qr * BT + b) * ET + e] = f2bf(o[nt][vg] / lrow[vg]);
    }
  }
}

// ---------------------------------------------------------------------------
// K3: out-projection.  A = ctx (bf16, async-staged), B = Wout (fp32 -> bf16).
// ---------------------------------------------------------------------------
__global__ void __launch_bounds__(256)
out_proj_kernel(const uint16_t* __restrict__ ctx, const float* __restrict__ W,
                const float* __restrict__ bias, float* __restrict__ out)
{
  __shared__ uint16_t As[BM * LDST];
  __shared__ uint16_t Bs[BN * LDST];

  const int m0 = blockIdx.x * BM;
  const int n0 = blockIdx.y * BN;
  const int tid  = threadIdx.x;
  const int lane = tid & 31;
  const int lo = lane & 15, hi = lane >> 4;
  const int wave = tid >> 5;
  const int wm = wave >> 1, wn = wave & 1;

  f32x8 acc[2][4];
  for (int i = 0; i < 2; ++i) for (int j = 0; j < 4; ++j) acc[i][j] = (f32x8)0.0f;

  const int srow = tid >> 1;
  const int scol = (tid & 1) * 16;

  for (int k0 = 0; k0 < ET; k0 += KC) {
    // stage A: raw bf16 -> async global->LDS copy (no VGPR round trip)
    async_copy16(As + srow * LDST + scol,
                 ctx + (size_t)(m0 + srow) * ET + k0 + scol);
    async_copy16(As + srow * LDST + scol + 8,
                 ctx + (size_t)(m0 + srow) * ET + k0 + scol + 8);
    { // stage B: fp32 -> bf16
      const float4* s4 = (const float4*)(W + (size_t)(n0 + srow) * ET + k0 + scol);
      uint4* dst = (uint4*)(Bs + srow * LDST + scol);
      dst[0] = cvt8(s4[0], s4[1]);
      dst[1] = cvt8(s4[2], s4[3]);
    }
    if (k0 + KC < ET) {
      __builtin_prefetch(ctx + (size_t)(m0 + srow) * ET + k0 + KC + scol, 0, 1);
      __builtin_prefetch(W + (size_t)(n0 + srow) * ET + k0 + KC + scol, 0, 1);
    }
    async_wait0();
    __syncthreads();

    Frag16 afr[2], bfr[4];
    for (int mt = 0; mt < 2; ++mt) {
      const uint16_t* ap = As + (wm * 32 + mt * 16 + lo) * LDST;
      afr[mt].q[0] = *(const uint4*)(ap + hi * 8);
      afr[mt].q[1] = *(const uint4*)(ap + 16 + hi * 8);
    }
    for (int nt = 0; nt < 4; ++nt) {
      const uint16_t* bp = Bs + (wn * 64 + nt * 16 + lo) * LDST + hi * 8;
      bfr[nt].q[0] = *(const uint4*)(bp);
      bfr[nt].q[1] = *(const uint4*)(bp + 16);
    }
    for (int nt = 0; nt < 4; ++nt)
      for (int mt = 0; mt < 2; ++mt)
        acc[mt][nt] = __builtin_amdgcn_wmma_f32_16x16x32_bf16(
            false, afr[mt].v, false, bfr[nt].v, (short)0, acc[mt][nt], false, false);
    __syncthreads();
  }

  for (int nt = 0; nt < 4; ++nt) {
    const int ncol = n0 + wn * 64 + nt * 16 + lo;
    const float bsum = bias[ncol];
    for (int mt = 0; mt < 2; ++mt)
      for (int vg = 0; vg < 8; ++vg) {
        const int mrow = m0 + wm * 32 + mt * 16 + vg + hi * 8;
        out[(size_t)mrow * ET + ncol] = acc[mt][nt][vg] + bsum;
      }
  }
}

// ---------------------------------------------------------------------------
extern "C" void kernel_launch(void* const* d_in, const int* in_sizes, int n_in,
                              void* d_out, int out_size, void* d_ws, size_t ws_size,
                              hipStream_t stream)
{
  const float* q    = (const float*)d_in[0];
  const float* k    = (const float*)d_in[1];
  const float* v    = (const float*)d_in[2];
  const unsigned char* mask = (const unsigned char*)d_in[3];
  const float* Win  = (const float*)d_in[4];
  const float* bin  = (const float*)d_in[5];
  const float* Wout = (const float*)d_in[6];
  const float* bout = (const float*)d_in[7];

  const size_t headElems = (size_t)BT * HT * LT * HD;
  uint16_t* qws = (uint16_t*)d_ws;
  uint16_t* kws = qws + headElems;
  uint16_t* vws = kws + headElems;
  uint16_t* ctx = vws + headElems;
  float* out = (float*)d_out;

  dim3 g1(LT * BT / BM, 3 * ET / BN);                   // 128 x 24
  in_proj_kernel<<<g1, 256, 0, stream>>>(q, k, v, Win, bin, qws, kws, vws);

  attn_kernel<<<BT * HT * 8, 128, 0, stream>>>(qws, kws, vws, mask, ctx);

  dim3 g3(LT * BT / BM, ET / BN);                       // 128 x 8
  out_proj_kernel<<<g3, 256, 0, stream>>>(ctx, Wout, bout, out);
}
